// MultiHeadSelfAttention_67156108640290
// MI455X (gfx1250) — matrix operand order repaired and checked by reading, not compiled, not run on hardware
//
#include <hip/hip_runtime.h>
#include <hip/hip_bf16.h>

typedef __attribute__((ext_vector_type(4)))  unsigned int u32x4;
typedef __attribute__((ext_vector_type(4)))  float        f32x4;
typedef __attribute__((ext_vector_type(16))) __bf16       v16bf;
typedef __attribute__((ext_vector_type(8)))  float        v8f;

union Frag { v16bf v; u32x4 q[2]; };

__device__ __forceinline__ unsigned short f2bf(float f) {
  unsigned int u = __float_as_uint(f);
  u += 0x7fffu + ((u >> 16) & 1u);           // round-to-nearest-even
  return (unsigned short)(u >> 16);
}
__device__ __forceinline__ unsigned int pk2bf(float lo, float hi) {
  return (unsigned int)f2bf(lo) | ((unsigned int)f2bf(hi) << 16);
}
__device__ __forceinline__ v8f wmma_bf16(const Frag& a, const Frag& b, v8f c) {
  return __builtin_amdgcn_wmma_f32_16x16x32_bf16(false, a.v, false, b.v,
                                                 (short)0, c, false, false);
}
// CDNA5 async global->LDS copy (tracked by ASYNCcnt)
__device__ __forceinline__ void async_copy_b128(unsigned lds_off, const void* gaddr) {
  asm volatile("global_load_async_to_lds_b128 %0, %1, off"
               :: "v"(lds_off), "v"(gaddr) : "memory");
}
__device__ __forceinline__ void wait_async0() {
  asm volatile("s_wait_asynccnt 0" ::: "memory");
}
__device__ __forceinline__ unsigned lds_addr(const void* p) {
  return (unsigned)(size_t)p;   // low 32 bits of generic addr = LDS offset
}

// ---------------------------------------------------------------------------
// Kernel 0a: f32 -> bf16, 8 elements/thread
// ---------------------------------------------------------------------------
__global__ __launch_bounds__(256) void cvt_bf16_kernel(
    const float* __restrict__ in, unsigned short* __restrict__ out, int n8) {
  const int i = blockIdx.x * 256 + threadIdx.x;
  if (i >= n8) return;
  const f32x4* p = (const f32x4*)(in + (size_t)i * 8);
  const f32x4 a = p[0], b = p[1];
  u32x4 o;
  o[0] = pk2bf(a[0], a[1]); o[1] = pk2bf(a[2], a[3]);
  o[2] = pk2bf(b[0], b[1]); o[3] = pk2bf(b[2], b[3]);
  *(u32x4*)(out + (size_t)i * 8) = o;
}

// ---------------------------------------------------------------------------
// Kernel 0b: weight (K x N, f32) -> transposed bf16 (N x K), LDS-tiled
// ---------------------------------------------------------------------------
__global__ __launch_bounds__(256) void transpose_cvt_kernel(
    const float* __restrict__ w, unsigned short* __restrict__ wt, int K, int N) {
  __shared__ unsigned short tile[32][33];
  const int k0 = blockIdx.y * 32, n0 = blockIdx.x * 32;
  const int tx = threadIdx.x & 31, ty = threadIdx.x >> 5;   // ty: 0..7
#pragma unroll
  for (int i = 0; i < 32; i += 8)
    tile[ty + i][tx] = f2bf(w[(size_t)(k0 + ty + i) * N + n0 + tx]);
  __syncthreads();
#pragma unroll
  for (int i = 0; i < 32; i += 8)
    wt[(size_t)(n0 + ty + i) * K + k0 + tx] = tile[tx][ty + i];
}

// ---------------------------------------------------------------------------
// Shared GEMM mainloop: C(128x64) += A(128xK) * B(Kx64), bf16 in / f32 acc.
// A: row-major bf16 (pitch Kdim); Bt: transposed bf16 [col][k] (pitch Kdim).
// 4 waves; wave tile 32x64; K-step 64; DOUBLE-BUFFERED async global->LDS
// copies (tile t+1 in flight while tile t feeds the WMMAs); one barrier/step.
// ---------------------------------------------------------------------------
__device__ __forceinline__ void issue_tile(
    const unsigned short* __restrict__ Abf, const unsigned short* __restrict__ Bt,
    unsigned short* As, unsigned short* Bs,
    int rowBase, int colBase, int Kdim, int k0, int tid) {
  { // A tile: 128 rows x 64 k  (1 row per thread, 8x b128)
    const unsigned short* src = Abf + (size_t)(rowBase + tid) * Kdim + k0;
    const unsigned lds = lds_addr(As + tid * 64);
#pragma unroll
    for (int j = 0; j < 8; ++j) async_copy_b128(lds + j * 16, src + j * 8);
  }
  { // B tile: 64 cols x 64 k (half row per thread, 4x b128)
    const int c = tid >> 1, kh = (tid & 1) * 32;
    const unsigned short* src = Bt + (size_t)(colBase + c) * Kdim + k0 + kh;
    const unsigned lds = lds_addr(Bs + c * 64 + kh);
#pragma unroll
    for (int j = 0; j < 4; ++j) async_copy_b128(lds + j * 16, src + j * 8);
  }
}

__device__ __forceinline__ void gemm_mainloop(
    const unsigned short* __restrict__ Abf, const unsigned short* __restrict__ Bt,
    unsigned short* As /*2 x 128*64*/, unsigned short* Bs /*2 x 64*64*/,
    int rowBase, int colBase, int Kdim, int tid, v8f acc[2][4]) {
  const int wave = tid >> 5, lane = tid & 31;
  const int half = lane >> 4, ln16 = lane & 15;

  issue_tile(Abf, Bt, As, Bs, rowBase, colBase, Kdim, 0, tid);   // prologue

  for (int k0 = 0; k0 < Kdim; k0 += 64) {
    const int cur = (k0 >> 6) & 1;
    unsigned short* Ac = As + cur * (128 * 64);
    unsigned short* Bc = Bs + cur * (64 * 64);

    wait_async0();       // own copies of tile k0 have landed
    __syncthreads();     // all waves' copies landed; prev reads of buf^1 done

    if (k0 + 64 < Kdim)  // stream tile k0+64 into the other buffer
      issue_tile(Abf, Bt, As + (cur ^ 1) * (128 * 64),
                 Bs + (cur ^ 1) * (64 * 64), rowBase, colBase, Kdim,
                 k0 + 64, tid);

#pragma unroll
    for (int f = 0; f < 2; ++f) {
      Frag a0, a1;
      const unsigned short* ar0 = Ac + (wave * 32 + ln16) * 64 + f * 32;
      const unsigned short* ar1 = ar0 + 16 * 64;
      a0.q[0] = *(const u32x4*)(ar0 + half * 8);
      a0.q[1] = *(const u32x4*)(ar0 + 16 + half * 8);
      a1.q[0] = *(const u32x4*)(ar1 + half * 8);
      a1.q[1] = *(const u32x4*)(ar1 + 16 + half * 8);
#pragma unroll
      for (int cg = 0; cg < 4; ++cg) {
        Frag b;
        const unsigned short* br = Bc + (cg * 16 + ln16) * 64 + f * 32 + half * 8;
        b.q[0] = *(const u32x4*)(br);
        b.q[1] = *(const u32x4*)(br + 16);
        acc[0][cg] = wmma_bf16(a0, b, acc[0][cg]);
        acc[1][cg] = wmma_bf16(a1, b, acc[1][cg]);
      }
    }
  }
}

// ---------------------------------------------------------------------------
// Kernel 1: qkv = xbf @ Wt ; scatter into Q (scaled), K, V^T (all bf16)
// ---------------------------------------------------------------------------
__global__ __launch_bounds__(128) void qkv_kernel(
    const unsigned short* __restrict__ xbf, const unsigned short* __restrict__ Wt,
    unsigned short* __restrict__ Qo, unsigned short* __restrict__ Ko,
    unsigned short* __restrict__ Vto) {
  constexpr int Kdim = 1024, N = 2048, H = 16, HD = 64;
  __shared__ unsigned short As[2 * 128 * 64];
  __shared__ unsigned short Bs[2 * 64 * 64];
  const int tid = threadIdx.x;
  const int wave = tid >> 5, lane = tid & 31;
  const int half = lane >> 4, ln16 = lane & 15;
  const int rowBase = blockIdx.y * 128;
  const int colBase = blockIdx.x * 64;

  v8f acc[2][4] = {{{}, {}, {}, {}}, {{}, {}, {}, {}}};
  gemm_mainloop(xbf, Wt, As, Bs, rowBase, colBase, Kdim, tid, acc);

#pragma unroll
  for (int rg = 0; rg < 2; ++rg)
#pragma unroll
    for (int cg = 0; cg < 4; ++cg) {
      const int col = colBase + cg * 16 + ln16;
      const int s = col >> 10;
      const int hh = (col & 1023) >> 6;
      const int e = col & 63;
#pragma unroll
      for (int r = 0; r < 8; ++r) {
        const int m = rowBase + wave * 32 + rg * 16 + half * 8 + r;
        const int n = m & (N - 1);
        const int bh = (m >> 11) * H + hh;
        const float v = acc[rg][cg][r];
        if (s == 0)
          Qo[((size_t)bh * N + n) * HD + e] = f2bf(v * 0.125f);
        else if (s == 1)
          Ko[((size_t)bh * N + n) * HD + e] = f2bf(v);
        else
          Vto[((size_t)bh * HD + e) * N + n] = f2bf(v);
      }
    }
}

// ---------------------------------------------------------------------------
// Kernel 2: flash attention per (b,h).  Block = 64 query rows (4 waves x 16).
// ---------------------------------------------------------------------------
__global__ __launch_bounds__(128) void attn_kernel(
    const unsigned short* __restrict__ Q, const unsigned short* __restrict__ K,
    const unsigned short* __restrict__ Vt, unsigned short* __restrict__ AO) {
  constexpr int N = 2048, HD = 64, C = 1024;
  const int bh = blockIdx.y;
  const int qBase = blockIdx.x * 64;
  const int tid = threadIdx.x;
  const int wave = tid >> 5, lane = tid & 31;
  const int half = lane >> 4, ln16 = lane & 15;

  __shared__ unsigned short Plds[4][16 * 64];

  const unsigned short* Qrow =
      Q + ((size_t)bh * N + qBase + wave * 16 + ln16) * HD;
  Frag qa[2];
#pragma unroll
  for (int f = 0; f < 2; ++f) {
    qa[f].q[0] = *(const u32x4*)(Qrow + f * 32 + half * 8);
    qa[f].q[1] = *(const u32x4*)(Qrow + f * 32 + 16 + half * 8);
  }

  v8f o[4] = {{}, {}, {}, {}};
  float mrow[8], lrow[8];
#pragma unroll
  for (int r = 0; r < 8; ++r) { mrow[r] = -3.0e38f; lrow[r] = 0.0f; }

  const unsigned short* Kbase = K + (size_t)bh * N * HD;
  const unsigned short* Vbase = Vt + (size_t)bh * HD * N;

  for (int kt = 0; kt < N; kt += 64) {
    if (kt + 64 < N)
      __builtin_prefetch(Kbase + (size_t)(kt + 64 + lane) * HD, 0, 0);

    // S = Q K^T
    v8f s[4] = {{}, {}, {}, {}};
#pragma unroll
    for (int g = 0; g < 4; ++g) {
      const unsigned short* Krow = Kbase + (size_t)(kt + g * 16 + ln16) * HD;
#pragma unroll
      for (int f = 0; f < 2; ++f) {
        Frag kb;
        kb.q[0] = *(const u32x4*)(Krow + f * 32 + half * 8);
        kb.q[1] = *(const u32x4*)(Krow + f * 32 + 16 + half * 8);
        s[g] = wmma_bf16(qa[f], kb, s[g]);
      }
    }
    // online softmax (row stats replicated across each 16-lane half)
#pragma unroll
    for (int r = 0; r < 8; ++r) {
      float mx = fmaxf(fmaxf(s[0][r], s[1][r]), fmaxf(s[2][r], s[3][r]));
#pragma unroll
      for (int off = 1; off < 16; off <<= 1)
        mx = fmaxf(mx, __shfl_xor(mx, off, 32));
      const float mnew = fmaxf(mrow[r], mx);
      const float alpha = __expf(mrow[r] - mnew);
      mrow[r] = mnew;
      float rsum = 0.0f;
#pragma unroll
      for (int g = 0; g < 4; ++g) {
        const float p = __expf(s[g][r] - mnew);
        s[g][r] = p;
        rsum += p;
      }
#pragma unroll
      for (int off = 1; off < 16; off <<= 1)
        rsum += __shfl_xor(rsum, off, 32);
      lrow[r] = lrow[r] * alpha + rsum;
#pragma unroll
      for (int g = 0; g < 4; ++g) o[g][r] *= alpha;
    }
    // P: C-layout -> A-layout via per-wave LDS bounce (wave LDS is in-order)
    unsigned short* Pw = Plds[wave];
#pragma unroll
    for (int g = 0; g < 4; ++g)
#pragma unroll
      for (int r = 0; r < 8; ++r)
        Pw[(half * 8 + r) * 64 + g * 16 + ln16] = f2bf(s[g][r]);
    asm volatile("s_wait_dscnt 0" ::: "memory");
    Frag pa[2];
#pragma unroll
    for (int f = 0; f < 2; ++f) {
      pa[f].q[0] = *(const u32x4*)(Pw + ln16 * 64 + f * 32 + half * 8);
      pa[f].q[1] = *(const u32x4*)(Pw + ln16 * 64 + f * 32 + 16 + half * 8);
    }
    // O += P V   (V transposed: contiguous B fragments)
#pragma unroll
    for (int cg = 0; cg < 4; ++cg) {
      const unsigned short* Vrow = Vbase + (size_t)(cg * 16 + ln16) * N + kt;
#pragma unroll
      for (int f = 0; f < 2; ++f) {
        Frag vb;
        vb.q[0] = *(const u32x4*)(Vrow + f * 32 + half * 8);
        vb.q[1] = *(const u32x4*)(Vrow + f * 32 + 16 + half * 8);
        o[cg] = wmma_bf16(pa[f], vb, o[cg]);
      }
    }
  }
  const int b = bh >> 4, h = bh & 15;
#pragma unroll
  for (int r = 0; r < 8; ++r) {
    const float inv = 1.0f / lrow[r];
    const int n = qBase + wave * 16 + half * 8 + r;
    const size_t rowoff = ((size_t)b * N + n) * C + h * HD;
#pragma unroll
    for (int cg = 0; cg < 4; ++cg)
      AO[rowoff + cg * 16 + ln16] = f2bf(o[cg][r] * inv);
  }
}

// ---------------------------------------------------------------------------
// Kernel 3: out = attn_out @ w_proj + b_proj   (f32 output)
// ---------------------------------------------------------------------------
__global__ __launch_bounds__(128) void proj_kernel(
    const unsigned short* __restrict__ AO, const unsigned short* __restrict__ Wt,
    const float* __restrict__ bias, float* __restrict__ out) {
  constexpr int Kdim = 1024, Nd = 1024;
  __shared__ unsigned short As[2 * 128 * 64];
  __shared__ unsigned short Bs[2 * 64 * 64];
  const int tid = threadIdx.x;
  const int wave = tid >> 5, lane = tid & 31;
  const int half = lane >> 4, ln16 = lane & 15;
  const int rowBase = blockIdx.y * 128;
  const int colBase = blockIdx.x * 64;

  v8f acc[2][4] = {{{}, {}, {}, {}}, {{}, {}, {}, {}}};
  gemm_mainloop(AO, Wt, As, Bs, rowBase, colBase, Kdim, tid, acc);

#pragma unroll
  for (int rg = 0; rg < 2; ++rg)
#pragma unroll
    for (int cg = 0; cg < 4; ++cg) {
      const int col = colBase + cg * 16 + ln16;
      const float bv = bias[col];
#pragma unroll
      for (int r = 0; r < 8; ++r) {
        const int m = rowBase + wave * 32 + rg * 16 + half * 8 + r;
        out[(size_t)m * Nd + col] = acc[rg][cg][r] + bv;
      }
    }
}

// ---------------------------------------------------------------------------
extern "C" void kernel_launch(void* const* d_in, const int* in_sizes, int n_in,
                              void* d_out, int out_size, void* d_ws,
                              size_t ws_size, hipStream_t stream) {
  (void)in_sizes; (void)n_in; (void)out_size; (void)ws_size;
  const float* x      = (const float*)d_in[0];
  const float* w_qkv  = (const float*)d_in[1];
  const float* w_proj = (const float*)d_in[2];
  const float* b_proj = (const float*)d_in[3];
  float* out = (float*)d_out;

  const size_t HE = (size_t)4 * 16 * 2048 * 64;   // 8,388,608
  unsigned short* Q      = (unsigned short*)d_ws;
  unsigned short* K      = Q + HE;
  unsigned short* Vt     = K + HE;
  unsigned short* XAO    = Vt + HE;               // x-bf16, later aliased as AO
  unsigned short* Wqkvt  = XAO + HE;              // 3072 x 1024
  unsigned short* Wprojt = Wqkvt + (size_t)3072 * 1024;  // 1024 x 1024

  // one-time precision/layout conversion
  cvt_bf16_kernel<<<dim3((int)(HE / 8 / 256)), 256, 0, stream>>>(x, XAO, (int)(HE / 8));
  transpose_cvt_kernel<<<dim3(96, 32), 256, 0, stream>>>(w_qkv, Wqkvt, 1024, 3072);
  transpose_cvt_kernel<<<dim3(32, 32), 256, 0, stream>>>(w_proj, Wprojt, 1024, 1024);

  qkv_kernel <<<dim3(48, 64), 128, 0, stream>>>(XAO, Wqkvt, Q, K, Vt);
  attn_kernel<<<dim3(32, 64), 128, 0, stream>>>(Q, K, Vt, XAO);   // AO overwrites xbf
  proj_kernel<<<dim3(16, 64), 128, 0, stream>>>(XAO, Wprojt, b_proj, out);
}
